// HeteroGNN_25211458027581
// MI455X (gfx1250) — compile-verified, matrix operand order checked by reading, not run on hardware
//
#include <hip/hip_runtime.h>

// ---------------- problem constants (match reference) ----------------
constexpr int HD   = 128;      // hidden channels
constexpr int FIN  = 64;       // transaction input features
constexpr int NT   = 500000;   // transaction nodes
constexpr int NC   = 100000;   // card nodes
constexpr int NE   = 80000;    // email nodes
constexpr int NEDGE= 500000;   // edges per relation

typedef __attribute__((ext_vector_type(16))) __bf16 v16bf;
typedef __attribute__((ext_vector_type(8)))  __bf16 v8bf;
typedef __attribute__((ext_vector_type(4)))  __bf16 v4bf;
typedef __attribute__((ext_vector_type(8)))  float  v8f;
typedef __attribute__((ext_vector_type(4)))  unsigned int v4u;
typedef __attribute__((ext_vector_type(8)))  int  v8i;
typedef __attribute__((ext_vector_type(4)))  int  v4i;

#if defined(__HIP_DEVICE_COMPILE__) && __has_builtin(__builtin_amdgcn_tensor_load_to_lds)
#define HAVE_TDM 1
#endif

__device__ __forceinline__ void atomAddF(float* p, float v) {
    unsafeAtomicAdd(p, v);   // global_atomic_add_f32
}

// load a 16x32 (or 32x16) bf16 WMMA fragment from padded LDS:
// lanes 0-15 hold K=base+0..7 and base+16..23 (two b128 loads), lanes 16-31 base=8.
__device__ __forceinline__ v16bf ldfrag(const __bf16* base) {
    union { v16bf v; v8bf h[2]; } u;
    u.h[0] = *(const v8bf*)(base);
    u.h[1] = *(const v8bf*)(base + 16);
    return u.v;
}

#ifdef HAVE_TDM
// ---- Tensor Data Mover: 2D bf16 tile load (tile 32 elems x 128 rows) into LDS
// with 16B padding after every 64B row -> row pitch 80B (APAD=40 bf16).
// D# layout per CDNA5 ISA.
__device__ __forceinline__ void tdm_load_tile_bf16(
    unsigned lds_off, const void* gptr, int tdim0, int tdim1, int stride_elems)
{
    unsigned long long ga = (unsigned long long)(uintptr_t)gptr & ((1ull << 57) - 1);
    v4u g0;
    g0[0] = 1u;                                   // count=1 (valid user descriptor)
    g0[1] = lds_off;                              // lds_addr (bytes)
    g0[2] = (unsigned)ga;                         // global_addr[31:0]
    g0[3] = (unsigned)(ga >> 32) | (2u << 30);    // global_addr[56:32] | type=2 ("image")

    // group1: data_size=1 (2B) @16, pad_enable @20,
    // pad_interval=3 (16 DWORDs = one 64B row) @22, pad_amount=3 (4 DWORDs = 16B) @25
    unsigned long long q0 = (unsigned long long)((1u << 16) | (1u << 20) | (3u << 22) | (3u << 25))
                          | ((unsigned long long)((unsigned)tdim0 & 0xFFFFu) << 48);  // tensor_dim0[15:0]
    unsigned long long q1 = (unsigned long long)((unsigned)tdim0 >> 16)               // tensor_dim0[31:16]
                          | ((unsigned long long)(unsigned)tdim1 << 16)               // tensor_dim1
                          | (32ull << 48);                                            // tile_dim0 = 32
    unsigned long long q2 = 128ull                                                    // tile_dim1 = 128
                          | ((unsigned long long)(unsigned)stride_elems << 32);       // dim0_stride[31:0]
    unsigned long long q3 = 0ull;
    v8i g1;
    g1[0] = (int)(unsigned)q0; g1[1] = (int)(q0 >> 32);
    g1[2] = (int)(unsigned)q1; g1[3] = (int)(q1 >> 32);
    g1[4] = (int)(unsigned)q2; g1[5] = (int)(q2 >> 32);
    g1[6] = (int)(unsigned)q3; g1[7] = (int)(q3 >> 32);
    v4i z4 = {};
#if __clang_major__ >= 23
    v8i z8 = {};
    __builtin_amdgcn_tensor_load_to_lds(g0, g1, z4, z4, z8, 0);
#else
    __builtin_amdgcn_tensor_load_to_lds(g0, g1, z4, z4, 0);
#endif
}
#endif

// ---------------- WMMA GEMM: out[N,128] = A[N,K] @ W[K,128] (+bias)(+acc)(relu) ----
// A: bf16 [N,K] row-major.  Wt: bf16 [128,K] row-major (pre-transposed weights).
template<bool ACC_IN, bool RELU, bool OUT_BF16>
__global__ __launch_bounds__(256) void gemm_h128(
    const __bf16* __restrict__ A, int N, int K,
    const __bf16* __restrict__ Wt, const float* __restrict__ bias,
    const float* __restrict__ acc_in, void* __restrict__ outraw)
{
    constexpr int APAD = 40;                  // row pitch in bf16 (80B: 16B aligned, bank-skewed)
    __shared__ __bf16 lA[2][128 * APAD];      // A tile  [m][k], double-buffered
    __shared__ __bf16 lB[2][128 * APAD];      // Wt tile [n][k], double-buffered

    const int tid  = threadIdx.x;
    const int lane = tid & 31;
    const int wave = tid >> 5;
    const int m0   = blockIdx.x * 128;
    const int nk   = K >> 5;                  // K is a multiple of 32

    v8f acc[8];
    #pragma unroll
    for (int i = 0; i < 8; i++) acc[i] = v8f{};

#ifdef HAVE_TDM
    // wave0 DMAs the A tile, wave1 DMAs the B tile; waves 2-7 only compute.
    auto issue = [&](int i) {
        const int kc = i << 5;
        if (wave == 0) {
            tdm_load_tile_bf16((unsigned)(uintptr_t)&lA[i & 1][0],
                               A + (size_t)m0 * K + kc, K - kc, N - m0, K);
        } else if (wave == 1) {
            tdm_load_tile_bf16((unsigned)(uintptr_t)&lB[i & 1][0],
                               Wt + kc, K - kc, 128, K);
        }
    };
    issue(0);
    for (int i = 0; i < nk; ++i) {
        if (i + 1 < nk) issue(i + 1);                 // prefetch next tile via TDM
        if (wave < 2) {
            if (i + 1 < nk) __builtin_amdgcn_s_wait_tensorcnt(1);  // tile i done, i+1 in flight
            else            __builtin_amdgcn_s_wait_tensorcnt(0);
        }
        __syncthreads();
        const __bf16* bufA = &lA[i & 1][0];
        const __bf16* bufB = &lB[i & 1][0];
        const int khalf = (lane < 16) ? 0 : 8;
        const v16bf afrag = ldfrag(bufA + (wave * 16 + (lane & 15)) * APAD + khalf);
        #pragma unroll
        for (int nt = 0; nt < 8; nt++) {
            const v16bf bfrag = ldfrag(bufB + (nt * 16 + (lane & 15)) * APAD + khalf);
            acc[nt] = __builtin_amdgcn_wmma_f32_16x16x32_bf16(
                false, afrag, false, bfrag, (short)0, acc[nt], false, false);
        }
        __syncthreads();
    }
#else
    // fallback: manual single-buffered staging (also what the host pass sees)
    for (int i = 0; i < nk; ++i) {
        const int kc = i << 5;
        {
            const int row  = tid >> 1;
            const int colh = (tid & 1) * 16;
            const int grow = m0 + row;
            __bf16* dst = &lA[0][row * APAD + colh];
            if (grow < N) {
                const __bf16* src = A + (size_t)grow * K + kc + colh;
                #pragma unroll
                for (int j = 0; j < 16; j++) dst[j] = src[j];
            } else {
                #pragma unroll
                for (int j = 0; j < 16; j++) dst[j] = (__bf16)0.f;
            }
            const __bf16* wsrc = Wt + (size_t)row * K + kc + colh;
            __bf16* wdst = &lB[0][row * APAD + colh];
            #pragma unroll
            for (int j = 0; j < 16; j++) wdst[j] = wsrc[j];
        }
        __syncthreads();
        const int khalf = (lane < 16) ? 0 : 8;
        const v16bf afrag = ldfrag(&lA[0][(wave * 16 + (lane & 15)) * APAD + khalf]);
        #pragma unroll
        for (int nt = 0; nt < 8; nt++) {
            const v16bf bfrag = ldfrag(&lB[0][(nt * 16 + (lane & 15)) * APAD + khalf]);
            acc[nt] = __builtin_amdgcn_wmma_f32_16x16x32_bf16(
                false, afrag, false, bfrag, (short)0, acc[nt], false, false);
        }
        __syncthreads();
    }
#endif

    // ---- epilogue: bias / fp32 accumulator add / relu / store
    const int col     = lane & 15;
    const int rowbase = m0 + wave * 16 + ((lane < 16) ? 0 : 8);
    #pragma unroll
    for (int nt = 0; nt < 8; nt++) {
        const int n = nt * 16 + col;
        const float b = bias ? bias[n] : 0.f;
        #pragma unroll
        for (int r = 0; r < 8; r++) {
            const int row = rowbase + r;
            if (row < N) {
                float v = acc[nt][r] + b;
                if (ACC_IN) v += acc_in[(size_t)row * HD + n];
                if (RELU)   v  = fmaxf(v, 0.f);
                if (OUT_BF16) ((__bf16*)outraw)[(size_t)row * HD + n] = (__bf16)v;
                else          ((float*) outraw)[(size_t)row * HD + n] = v;
            }
        }
    }
}

// ---------------- conversion kernels ----------------
__global__ __launch_bounds__(256) void k_cvt_bf16(const float* __restrict__ in,
                                                  __bf16* __restrict__ out, int n) {
    int i = blockIdx.x * blockDim.x + threadIdx.x;
    if (i < n) out[i] = (__bf16)in[i];
}

// Wt[n*K + k] = W[k*HD + n]  (transpose + bf16), W is [K, HD]
__global__ __launch_bounds__(256) void k_cvtWt(const float* __restrict__ W, int K,
                                               __bf16* __restrict__ Wt) {
    int i = blockIdx.x * blockDim.x + threadIdx.x;
    if (i < K * HD) {
        int n = i / K, k = i - n * K;
        Wt[i] = (__bf16)W[k * HD + n];
    }
}

// ---------------- graph kernels ----------------
__global__ __launch_bounds__(256) void k_deg(const int* __restrict__ idx, int E,
                                             float* __restrict__ deg) {
    int i = blockIdx.x * blockDim.x + threadIdx.x;
    if (i < E) atomAddF(&deg[idx[i]], 1.0f);
}

// one wave per edge: gather bf16 row at src, fp32 atomic-accumulate at dst
__global__ __launch_bounds__(256) void k_scatter(const __bf16* __restrict__ x,
                                                 const int* __restrict__ src,
                                                 const int* __restrict__ dst, int E,
                                                 float* __restrict__ agg) {
    const int gw   = (blockIdx.x * blockDim.x + threadIdx.x) >> 5;
    const int lane = threadIdx.x & 31;
    if (gw >= E) return;
    const int s = src[gw];
    const int d = dst[gw];
    const __bf16* xs = x + (size_t)s * HD;
    float* ad = agg + (size_t)d * HD;
    __builtin_prefetch(ad + lane * 4, 0, 0);
    const v4bf v = ((const v4bf*)xs)[lane];
    #pragma unroll
    for (int i = 0; i < 4; i++) atomAddF(ad + lane * 4 + i, (float)v[i]);
}

// mean-normalize fp32 accumulator and emit bf16 matrix for the WMMA GEMMs
__global__ __launch_bounds__(256) void k_mean(const float* __restrict__ agg,
                                              const float* __restrict__ deg,
                                              __bf16* __restrict__ outb, int n) {
    int i = blockIdx.x * blockDim.x + threadIdx.x;
    if (i < n * HD) {
        const float c = deg[i >> 7];           // / HD
        outb[i] = (__bf16)(agg[i] / fmaxf(c, 1.0f));
    }
}

__global__ void k_bsum(const float* __restrict__ a, const float* __restrict__ b,
                       float* __restrict__ o) {
    int i = threadIdx.x;
    if (i < HD) o[i] = a[i] + b[i];
}

// classifier: out[i] = dot(t[i,:], w) + b ; one wave per row
__global__ __launch_bounds__(256) void k_cls(const __bf16* __restrict__ t,
                                             const float* __restrict__ w,
                                             const float* __restrict__ b,
                                             float* __restrict__ out, int N) {
    const int gw   = (blockIdx.x * blockDim.x + threadIdx.x) >> 5;
    const int lane = threadIdx.x & 31;
    if (gw >= N) return;
    const __bf16* row = t + (size_t)gw * HD;
    float s = 0.f;
    #pragma unroll
    for (int i = 0; i < 4; i++) s += (float)row[lane + i * 32] * w[lane + i * 32];
    #pragma unroll
    for (int off = 16; off > 0; off >>= 1) s += __shfl_down(s, off, 32);
    if (lane == 0) out[gw] = s + b[0];
}

// ---------------- host orchestration ----------------
extern "C" void kernel_launch(void* const* d_in, const int* in_sizes, int n_in,
                              void* d_out, int out_size, void* d_ws, size_t ws_size,
                              hipStream_t stream) {
    (void)in_sizes; (void)n_in; (void)out_size; (void)ws_size;
    const float* x_t   = (const float*)d_in[0];
    const int* uc_src  = (const int*)d_in[1];
    const int* uc_dst  = (const int*)d_in[2];
    const int* he_src  = (const int*)d_in[3];
    const int* he_dst  = (const int*)d_in[4];
    const float* tlinW = (const float*)d_in[5];
    const float* tlinB = (const float*)d_in[6];
    const float* c1Wl  = (const float*)d_in[7];
    const float* c1bl  = (const float*)d_in[8];
    const float* c1Wr  = (const float*)d_in[9];
    const float* c2Wl  = (const float*)d_in[10];
    const float* c2bl  = (const float*)d_in[11];
    const float* c2Wr  = (const float*)d_in[12];
    const float* clsW  = (const float*)d_in[13];
    const float* clsB  = (const float*)d_in[14];
    float* out = (float*)d_out;

    // workspace carve (256B aligned)
    char* p = (char*)d_ws;
    auto take = [&](size_t bytes) { char* r = p; p += (bytes + 255) & ~(size_t)255; return r; };
    __bf16* xbf   = (__bf16*)take((size_t)NT * FIN * 2);  // bf16 copy of x_transaction
    __bf16* xtA   = (__bf16*)take((size_t)NT * HD * 2);   // t features ping
    __bf16* xtB   = (__bf16*)take((size_t)NT * HD * 2);   // t features pong
    __bf16* xcB   = (__bf16*)take((size_t)NC * HD * 2);   // layer-1 card features
    __bf16* xeB   = (__bf16*)take((size_t)NE * HD * 2);   // layer-1 email features
    float*  agg   = (float*) take((size_t)NT * HD * 4);   // scatter fp32 accumulator
    __bf16* aggB  = (__bf16*)take((size_t)NT * HD * 2);   // mean result, bf16 (GEMM A)
    float*  acc   = (float*) take((size_t)NT * HD * 4);   // fp32 GEMM chain accumulator
    float*  degC  = (float*) take((size_t)NC * 4);
    float*  degE  = (float*) take((size_t)NE * 4);
    float*  degTu = (float*) take((size_t)NT * 4);
    float*  degTh = (float*) take((size_t)NT * 4);
    float*  bsum  = (float*) take((size_t)HD * 4);
    // pre-transposed bf16 weights Wt[n][k]
    __bf16* wtTlin = (__bf16*)take((size_t)HD * FIN * 2);
    __bf16* wt1l0  = (__bf16*)take((size_t)HD * HD * 2);
    __bf16* wt1l2  = (__bf16*)take((size_t)HD * HD * 2);
    __bf16* wt1r1  = (__bf16*)take((size_t)HD * HD * 2);
    __bf16* wt1r3  = (__bf16*)take((size_t)HD * HD * 2);
    __bf16* wt2l1  = (__bf16*)take((size_t)HD * HD * 2);
    __bf16* wt2l3  = (__bf16*)take((size_t)HD * HD * 2);
    __bf16* wt2r1  = (__bf16*)take((size_t)HD * HD * 2);
    __bf16* wt2r3  = (__bf16*)take((size_t)HD * HD * 2);

    const dim3 blk(256);
    const int edgeBlocks = (NEDGE * 32 + 255) / 256;   // one wave per edge
    auto gemmGrid = [](int n) { return dim3((n + 127) / 128); };
    const int wtBlocks = (HD * HD + 255) / 256;

    // ---- one-time conversions (per launch; deterministic)
    k_cvt_bf16<<<(NT * FIN + 255) / 256, blk, 0, stream>>>(x_t, xbf, NT * FIN);
    k_cvtWt<<<(HD * FIN + 255) / 256, blk, 0, stream>>>(tlinW, FIN, wtTlin);
    k_cvtWt<<<wtBlocks, blk, 0, stream>>>(c1Wl + 0 * HD * HD, HD, wt1l0);
    k_cvtWt<<<wtBlocks, blk, 0, stream>>>(c1Wl + 2 * HD * HD, HD, wt1l2);
    k_cvtWt<<<wtBlocks, blk, 0, stream>>>(c1Wr + 1 * HD * HD, HD, wt1r1);
    k_cvtWt<<<wtBlocks, blk, 0, stream>>>(c1Wr + 3 * HD * HD, HD, wt1r3);
    k_cvtWt<<<wtBlocks, blk, 0, stream>>>(c2Wl + 1 * HD * HD, HD, wt2l1);
    k_cvtWt<<<wtBlocks, blk, 0, stream>>>(c2Wl + 3 * HD * HD, HD, wt2l3);
    k_cvtWt<<<wtBlocks, blk, 0, stream>>>(c2Wr + 1 * HD * HD, HD, wt2r1);
    k_cvtWt<<<wtBlocks, blk, 0, stream>>>(c2Wr + 3 * HD * HD, HD, wt2r3);

    // ---- degrees (fixed per relation/direction, reused by both layers)
    hipMemsetAsync(degC,  0, (size_t)NC * 4, stream);
    hipMemsetAsync(degE,  0, (size_t)NE * 4, stream);
    hipMemsetAsync(degTu, 0, (size_t)NT * 4, stream);
    hipMemsetAsync(degTh, 0, (size_t)NT * 4, stream);
    k_deg<<<(NEDGE + 255) / 256, blk, 0, stream>>>(uc_dst, NEDGE, degC);
    k_deg<<<(NEDGE + 255) / 256, blk, 0, stream>>>(uc_src, NEDGE, degTu);
    k_deg<<<(NEDGE + 255) / 256, blk, 0, stream>>>(he_dst, NEDGE, degE);
    k_deg<<<(NEDGE + 255) / 256, blk, 0, stream>>>(he_src, NEDGE, degTh);
    k_bsum<<<1, 128, 0, stream>>>(c1bl + 1 * HD, c1bl + 3 * HD, bsum);

    // ---- xt = x @ tlin_W + tlin_b   (K=64 -> bf16 out)
    gemm_h128<false,false,true><<<gemmGrid(NT), blk, 0, stream>>>(
        xbf, NT, FIN, wtTlin, tlinB, nullptr, xtA);

    // ================= layer 1 (xc = xe = 0 exploited exactly) =================
    // c1 = relu(mean_{uc: t->c}(xt) @ Wl0 + bl0)
    hipMemsetAsync(agg, 0, (size_t)NC * HD * 4, stream);
    k_scatter<<<edgeBlocks, blk, 0, stream>>>(xtA, uc_src, uc_dst, NEDGE, agg);
    k_mean<<<(NC * HD + 255) / 256, blk, 0, stream>>>(agg, degC, aggB, NC);
    gemm_h128<false,true,true><<<gemmGrid(NC), blk, 0, stream>>>(
        aggB, NC, HD, wt1l0, c1bl + 0 * HD, nullptr, xcB);

    // e1 = relu(mean_{he: t->e}(xt) @ Wl2 + bl2)
    hipMemsetAsync(agg, 0, (size_t)NE * HD * 4, stream);
    k_scatter<<<edgeBlocks, blk, 0, stream>>>(xtA, he_src, he_dst, NEDGE, agg);
    k_mean<<<(NE * HD + 255) / 256, blk, 0, stream>>>(agg, degE, aggB, NE);
    gemm_h128<false,true,true><<<gemmGrid(NE), blk, 0, stream>>>(
        aggB, NE, HD, wt1l2, c1bl + 2 * HD, nullptr, xeB);

    // t1 = relu(xt@Wr1 + xt@Wr3 + bl1 + bl3)  (c/e inputs are zero in layer 1)
    gemm_h128<false,false,false><<<gemmGrid(NT), blk, 0, stream>>>(
        xtA, NT, HD, wt1r1, bsum, nullptr, acc);
    gemm_h128<true,true,true><<<gemmGrid(NT), blk, 0, stream>>>(
        xtA, NT, HD, wt1r3, nullptr, acc, xtB);

    // ================= layer 2 (only t output feeds the classifier) ============
    // acc  = mean_{uc: c->t}(c1) @ Wl1 + bl1
    hipMemsetAsync(agg, 0, (size_t)NT * HD * 4, stream);
    k_scatter<<<edgeBlocks, blk, 0, stream>>>(xcB, uc_dst, uc_src, NEDGE, agg);
    k_mean<<<(NT * HD + 255) / 256, blk, 0, stream>>>(agg, degTu, aggB, NT);
    gemm_h128<false,false,false><<<gemmGrid(NT), blk, 0, stream>>>(
        aggB, NT, HD, wt2l1, c2bl + 1 * HD, nullptr, acc);

    // acc += mean_{he: e->t}(e1) @ Wl3 + bl3
    hipMemsetAsync(agg, 0, (size_t)NT * HD * 4, stream);
    k_scatter<<<edgeBlocks, blk, 0, stream>>>(xeB, he_dst, he_src, NEDGE, agg);
    k_mean<<<(NT * HD + 255) / 256, blk, 0, stream>>>(agg, degTh, aggB, NT);
    gemm_h128<true,false,false><<<gemmGrid(NT), blk, 0, stream>>>(
        aggB, NT, HD, wt2l3, c2bl + 3 * HD, acc, acc);

    // acc += t1@Wr1 ; t2 = relu(acc + t1@Wr3)
    gemm_h128<true,false,false><<<gemmGrid(NT), blk, 0, stream>>>(
        xtB, NT, HD, wt2r1, nullptr, acc, acc);
    gemm_h128<true,true,true><<<gemmGrid(NT), blk, 0, stream>>>(
        xtB, NT, HD, wt2r3, nullptr, acc, xtA);

    // ---- classifier
    k_cls<<<(NT * 32 + 255) / 256, blk, 0, stream>>>(xtA, clsW, clsB, out, NT);
}
